// GAT_28149215658569
// MI455X (gfx1250) — matrix operand order repaired and checked by reading, not compiled, not run on hardware
//
#include <hip/hip_runtime.h>
#include <cstdint>
#include <cstddef>

// ---------------------------------------------------------------------------
// Types for CDNA5 WMMA (wave32): 16x16x32 bf16 -> f32
// ---------------------------------------------------------------------------
typedef __bf16 bf16;
typedef __attribute__((ext_vector_type(16))) __bf16 v16bf;
typedef __attribute__((ext_vector_type(8)))  float  v8f;

union FragAB { v16bf v; uint4 q[2]; };

#define BM 128
#define BN 128
#define BK 32
#define LDT 40   // padded LDS row stride (elements); keeps 16B alignment, breaks bank conflicts

__device__ __forceinline__ float sigm(float x) { return 1.f / (1.f + expf(-x)); }

// ---------------------------------------------------------------------------
// Tensor Data Mover staging (gfx1250): one wave issues a 2D tile DMA
// (global -> LDS) described by D# group0/group1; tracked on TENSORcnt.
// Falls back to manual global->VGPR->LDS staging if the builtin is absent.
// ---------------------------------------------------------------------------
#if __has_builtin(__builtin_amdgcn_tensor_load_to_lds) && \
    __has_builtin(__builtin_amdgcn_s_wait_tensorcnt)
#define HAVE_TDM 1
#else
#define HAVE_TDM 0
#endif

#if HAVE_TDM
typedef __attribute__((ext_vector_type(4))) unsigned int u32x4;
typedef __attribute__((ext_vector_type(4))) int          i32x4;
typedef __attribute__((ext_vector_type(8))) int          i32x8;

__device__ __forceinline__ unsigned lds_off(const void* p) {
    // generic -> LDS address space, then to raw LDS byte offset
    return (unsigned)(size_t)(__attribute__((address_space(3))) const char*)p;
}

// Load a (tile_rows x BK) bf16 tile whose rows are K elements apart in global
// memory into LDS rows padded to LDT elements. Rows >= rows_rem are zero-filled
// by the TDM (tensor_dim1 OOB). LDS padding: after every 16 dwords (64B row),
// skip 4 dwords (16B) => row stride 80B = LDT bf16.
__device__ __forceinline__ void tdm_load_tile(unsigned lds, unsigned long long ga,
                                              unsigned k_elems, unsigned rows_rem,
                                              unsigned tile_rows) {
    u32x4 g0;
    g0[0] = 1u;                                                     // count=1, user D#
    g0[1] = lds;                                                    // lds_addr
    g0[2] = (unsigned)ga;                                           // global_addr[31:0]
    g0[3] = (unsigned)((ga >> 32) & 0x1FFFFFFull) | (2u << 30);     // ga[56:32] | type=2
    i32x8 g1;
    g1[0] = (int)((1u << 16) |            // data_size = 2 bytes
                  (1u << 20) |            // pad_enable
                  (3u << 22) |            // pad_interval: 16 dwords
                  (3u << 25));            // pad_amount:   4 dwords
    g1[1] = (int)((k_elems & 0xFFFFu) << 16);                       // tensor_dim0 lo
    g1[2] = (int)((k_elems >> 16) | ((rows_rem & 0xFFFFu) << 16));  // dim0 hi | dim1 lo
    g1[3] = (int)((rows_rem >> 16) | ((unsigned)BK << 16));         // dim1 hi | tile_dim0
    g1[4] = (int)tile_rows;                                         // tile_dim1 (tile_dim2=0)
    g1[5] = (int)k_elems;                                           // tensor_dim0_stride lo
    g1[6] = 0;                                                      // stride0 hi | stride1 lo
    g1[7] = 0;                                                      // stride1 hi
    i32x4 z4 = {0, 0, 0, 0};
#if defined(__clang_major__) && (__clang_major__ >= 23)
    i32x8 z8 = {0, 0, 0, 0, 0, 0, 0, 0};
    __builtin_amdgcn_tensor_load_to_lds(g0, g1, z4, z4, z8, 0);
#else
    __builtin_amdgcn_tensor_load_to_lds(g0, g1, z4, z4, 0);
#endif
}
#endif  // HAVE_TDM

// ---------------------------------------------------------------------------
// WMMA GEMM:  C[M,N] = A[M,K] (bf16, row-major) x W[N,K]^T (bf16, row-major)
// MODE 0: Cf = acc                       (f32 out)
// MODE 1: Cb = bf16(acc)                 (bf16 out, feeds next GEMM)
// MODE 2: Cf = sigmoid(acc + bias[n]) * f32(aux)            (gate pass 1)
// MODE 3: Cb = bf16(Cf + sigmoid(acc + bias[n]) * f32(aux)) (gate pass 2)
// ---------------------------------------------------------------------------
template <int MODE>
__global__ __launch_bounds__(256) void k_gemm(
    const bf16* __restrict__ A, const bf16* __restrict__ W,
    float* __restrict__ Cf, bf16* __restrict__ Cb,
    const bf16* __restrict__ aux, const float* __restrict__ bias,
    int M, int N, int K)
{
    __shared__ bf16 As[2][BM * LDT];
    __shared__ bf16 Bs[2][BN * LDT];

    const int t     = threadIdx.x;
    const int wid   = t >> 5;        // 8 waves
    const int lane  = t & 31;
    const int hlf   = lane >> 4;     // 0/1: lane half
    const int r     = lane & 15;
    const int waveM = wid >> 1;      // 0..3  -> 32-row strips
    const int waveN = wid & 1;       // 0..1  -> 64-col strips
    const int blockM = blockIdx.y * BM;
    const int blockN = blockIdx.x * BN;

    v8f acc[2][4];
    for (int i = 0; i < 2; i++)
        for (int j = 0; j < 4; j++)
            for (int v = 0; v < 8; v++) acc[i][j][v] = 0.f;

#if HAVE_TDM
    // ---- TDM double-buffered pipeline: DMA tile k+1 while computing tile k ----
    const unsigned long long gaA = (unsigned long long)(size_t)A + 2ull * ((size_t)blockM * K);
    const unsigned long long gaB = (unsigned long long)(size_t)W + 2ull * ((size_t)blockN * K);
    const unsigned rowsA = (unsigned)(M - blockM);
    const unsigned rowsB = (unsigned)(N - blockN);
    int p = 0;
    if (wid == 0) {
        tdm_load_tile(lds_off(&As[0][0]), gaA, (unsigned)K, rowsA, BM);
        tdm_load_tile(lds_off(&Bs[0][0]), gaB, (unsigned)K, rowsB, BN);
    }
    for (int k0 = 0; k0 < K; k0 += BK) {
        if (wid == 0) __builtin_amdgcn_s_wait_tensorcnt(0);
        __syncthreads();                       // buffer p ready; buffer p^1 free
        if (wid == 0 && (k0 + BK) < K) {       // prefetch next tile via TDM
            tdm_load_tile(lds_off(&As[p ^ 1][0]), gaA + 2ull * (k0 + BK),
                          (unsigned)K, rowsA, BM);
            tdm_load_tile(lds_off(&Bs[p ^ 1][0]), gaB + 2ull * (k0 + BK),
                          (unsigned)K, rowsB, BN);
        }
        FragAB a[2], b[4];
        for (int i = 0; i < 2; i++) {
            const uint4* pa = (const uint4*)&As[p][(waveM * 32 + i * 16 + r) * LDT + 16 * hlf];
            a[i].q[0] = pa[0];
            a[i].q[1] = pa[1];
        }
        for (int j = 0; j < 4; j++) {
            const uint4* pb = (const uint4*)&Bs[p][(waveN * 64 + j * 16 + r) * LDT + 16 * hlf];
            b[j].q[0] = pb[0];
            b[j].q[1] = pb[1];
        }
        for (int i = 0; i < 2; i++)
            for (int j = 0; j < 4; j++)
                acc[i][j] = __builtin_amdgcn_wmma_f32_16x16x32_bf16(
                    false, a[i].v, false, b[j].v, (short)0, acc[i][j], false, false);
        p ^= 1;
    }
#else
    // ---- fallback: manual global->VGPR->LDS staging ----
    const int lrow = t >> 1;          // 0..127
    const int lcol = (t & 1) << 4;    // 0 or 16 (bf16 elements)
    for (int k0 = 0; k0 < K; k0 += BK) {
        __syncthreads();
        {
            uint4 d0 = {0, 0, 0, 0}, d1 = {0, 0, 0, 0};
            const int gm = blockM + lrow;
            if (gm < M) {
                const uint4* p = (const uint4*)(A + (size_t)gm * K + k0 + lcol);
                d0 = p[0]; d1 = p[1];
            }
            uint4* s = (uint4*)&As[0][lrow * LDT + lcol];
            s[0] = d0; s[1] = d1;
        }
        {
            const uint4* p = (const uint4*)(W + (size_t)(blockN + lrow) * K + k0 + lcol);
            const uint4 d0 = p[0], d1 = p[1];
            uint4* s = (uint4*)&Bs[0][lrow * LDT + lcol];
            s[0] = d0; s[1] = d1;
        }
        __syncthreads();
        FragAB a[2], b[4];
        for (int i = 0; i < 2; i++) {
            const uint4* pa = (const uint4*)&As[0][(waveM * 32 + i * 16 + r) * LDT + 16 * hlf];
            a[i].q[0] = pa[0];
            a[i].q[1] = pa[1];
        }
        for (int j = 0; j < 4; j++) {
            const uint4* pb = (const uint4*)&Bs[0][(waveN * 64 + j * 16 + r) * LDT + 16 * hlf];
            b[j].q[0] = pb[0];
            b[j].q[1] = pb[1];
        }
        for (int i = 0; i < 2; i++)
            for (int j = 0; j < 4; j++)
                acc[i][j] = __builtin_amdgcn_wmma_f32_16x16x32_bf16(
                    false, a[i].v, false, b[j].v, (short)0, acc[i][j], false, false);
    }
#endif

    // ---- epilogue: C element (m = base + 8*hlf + v, n = base + r) = acc[v] ----
    for (int i = 0; i < 2; i++) {
        for (int j = 0; j < 4; j++) {
            const int ng = blockN + waveN * 64 + j * 16 + r;
            if (ng >= N) continue;
            const int   mbase = blockM + waveM * 32 + i * 16 + hlf * 8;
            const float bvl   = (MODE >= 2) ? bias[ng] : 0.f;
            for (int v = 0; v < 8; v++) {
                const int m = mbase + v;
                if (m >= M) break;
                const size_t idx = (size_t)m * N + ng;
                const float  x   = acc[i][j][v];
                if (MODE == 0) {
                    Cf[idx] = x;
                } else if (MODE == 1) {
                    Cb[idx] = (bf16)x;
                } else if (MODE == 2) {
                    Cf[idx] = sigm(x + bvl) * (float)aux[idx];
                } else {
                    const float f = Cf[idx] + sigm(x + bvl) * (float)aux[idx];
                    Cb[idx] = (bf16)f;
                }
            }
        }
    }
}

// ---------------------------------------------------------------------------
// Elementwise / graph kernels
// ---------------------------------------------------------------------------
__global__ void k_cvt_bf16(const float* __restrict__ s, bf16* __restrict__ d, size_t n) {
    const size_t i = (size_t)blockIdx.x * blockDim.x + threadIdx.x;
    if (i < n) d[i] = (bf16)s[i];
}

// dst[c*R + r] = src[r*C + c]  (f32 -> bf16 transpose, for Wv / We)
__global__ void k_transpose_bf16(const float* __restrict__ s, bf16* __restrict__ d, int R, int C) {
    const int i = blockIdx.x * blockDim.x + threadIdx.x;
    if (i >= R * C) return;
    const int rr = i / C, cc = i % C;
    d[(size_t)cc * R + rr] = (bf16)s[(size_t)rr * C + cc];
}

__global__ void k_fill(float* __restrict__ p, float v, size_t n) {
    const size_t i = (size_t)blockIdx.x * blockDim.x + threadIdx.x;
    if (i < n) p[i] = v;
}

// s_src[n,h] = z[n,h,:] . a[h,:O] ; s_dst[n,h] = z[n,h,:] . a[h,O:]
__global__ void k_scores(const float* __restrict__ z, const float* __restrict__ a,
                         float* __restrict__ ss, float* __restrict__ sd,
                         int Nn, int H, int O) {
    const int idx = blockIdx.x * blockDim.x + threadIdx.x;
    if (idx >= Nn * H) return;
    const int n = idx / H, h = idx % H;
    const float* zp = z + (size_t)n * H * O + (size_t)h * O;
    const float* ap = a + (size_t)h * 2 * O;
    float s0 = 0.f, s1 = 0.f;
    for (int o = 0; o < O; o++) { s0 += zp[o] * ap[o]; s1 += zp[o] * ap[O + o]; }
    ss[idx] = s0;
    sd[idx] = s1;
}

__device__ __forceinline__ void atomicMaxF(float* addr, float val) {
    if (val >= 0.f) atomicMax((int*)addr, __float_as_int(val));
    else            atomicMin((unsigned int*)addr, __float_as_uint(val));
}

// pass 1: e = leakyrelu(s_src[src]+s_dst[dst]); segment max via float atomic-max
__global__ void k_edge_max(const int* __restrict__ src, const int* __restrict__ dst,
                           const float* __restrict__ ss, const float* __restrict__ sd,
                           float* __restrict__ ebuf, float* __restrict__ m,
                           int E, int H) {
    const int idx = blockIdx.x * blockDim.x + threadIdx.x;
    if (idx >= E * H) return;
    const int e = idx / H, h = idx % H;
    float v = ss[(size_t)src[e] * H + h] + sd[(size_t)dst[e] * H + h];
    v = (v > 0.f) ? v : 0.01f * v;
    ebuf[idx] = v;
    atomicMaxF(&m[(size_t)dst[e] * H + h], v);
}

// pass 2: ex = exp(e - m[dst]); den[dst] += ex
__global__ void k_edge_exp(const int* __restrict__ dst, float* __restrict__ ebuf,
                           const float* __restrict__ m, float* __restrict__ den,
                           int E, int H) {
    const int idx = blockIdx.x * blockDim.x + threadIdx.x;
    if (idx >= E * H) return;
    const int e = idx / H, h = idx % H;
    const float ex = expf(ebuf[idx] - m[(size_t)dst[e] * H + h]);
    ebuf[idx] = ex;
    atomicAdd(&den[(size_t)dst[e] * H + h], ex);
}

// pass 3: h[dst] += (ex/den[dst]) * z[src]   (one block per edge)
__global__ void k_edge_aggr(const int* __restrict__ src, const int* __restrict__ dst,
                            const float* __restrict__ ebuf, const float* __restrict__ den,
                            const float* __restrict__ z, float* __restrict__ hout,
                            int E, int H, int O) {
    const int e = blockIdx.x;
    if (e >= E) return;
    const int s = src[e], d = dst[e];
    const int HO = H * O;
    for (int c = threadIdx.x; c < HO; c += blockDim.x) {
        const int   h     = c / O;
        const float alpha = ebuf[(size_t)e * H + h] / den[(size_t)d * H + h];
        atomicAdd(&hout[(size_t)d * HO + c], alpha * z[(size_t)s * HO + c]);
    }
}

__global__ void k_elu_bf16(const float* __restrict__ h, bf16* __restrict__ o, size_t n) {
    const size_t i = (size_t)blockIdx.x * blockDim.x + threadIdx.x;
    if (i >= n) return;
    const float x = h[i];
    o[i] = (bf16)((x > 0.f) ? x : (expf(x) - 1.f));
}

// ---------------------------------------------------------------------------
// Host launcher
// ---------------------------------------------------------------------------
extern "C" void kernel_launch(void* const* d_in, const int* in_sizes, int n_in,
                              void* d_out, int out_size, void* d_ws, size_t ws_size,
                              hipStream_t stream) {
    (void)n_in; (void)ws_size;
    const float* img   = (const float*)d_in[0];
    const float* blk   = (const float*)d_in[1];
    const float* Wimg  = (const float*)d_in[2];
    const float* Wblk  = (const float*)d_in[3];
    const float* Wv    = (const float*)d_in[4];
    const float* bv    = (const float*)d_in[5];
    const float* We    = (const float*)d_in[6];
    const float* be    = (const float*)d_in[7];
    const float* fc1   = (const float*)d_in[8];
    const float* attn1 = (const float*)d_in[9];
    const float* fc2   = (const float*)d_in[10];
    const float* attn2 = (const float*)d_in[11];
    const int*   e0s   = (const int*)d_in[12];
    const int*   e0d   = (const int*)d_in[13];
    const int*   e1s   = (const int*)d_in[14];
    const int*   e1d   = (const int*)d_in[15];
    const int E0 = in_sizes[12];
    const int E1 = in_sizes[14];

    constexpr int NS = 60000, ND0 = 30000, ND1 = 8000;
    constexpr int IMGD = 1024, BLKD = 768, F = 512;
    constexpr int H1 = 8, O1 = 64, O2 = 128;

    // ---- workspace bump allocator (256B aligned) ----
    uint8_t* ws  = (uint8_t*)d_ws;
    size_t   off = 0;
    auto alloc = [&](size_t bytes) -> void* {
        void* p = ws + off;
        off += (bytes + 255) & ~(size_t)255;
        return p;
    };
    bf16*  Ximg   = (bf16*)alloc((size_t)NS * IMGD * 2);
    bf16*  Xblk   = (bf16*)alloc((size_t)NS * BLKD * 2);
    bf16*  Wimgb  = (bf16*)alloc((size_t)F * IMGD * 2);
    bf16*  Wblkb  = (bf16*)alloc((size_t)F * BLKD * 2);
    bf16*  WvTb   = (bf16*)alloc((size_t)F * F * 2);
    bf16*  WeTb   = (bf16*)alloc((size_t)F * F * 2);
    bf16*  fc1b   = (bf16*)alloc((size_t)F * F * 2);
    bf16*  fc2b   = (bf16*)alloc((size_t)O2 * F * 2);
    bf16*  fib    = (bf16*)alloc((size_t)NS * F * 2);
    bf16*  tib    = (bf16*)alloc((size_t)NS * F * 2);
    float* fusedF = (float*)alloc((size_t)NS * F * 4);
    bf16*  fusedB = (bf16*)alloc((size_t)NS * F * 2);
    float* z1     = (float*)alloc((size_t)NS * F * 4);
    float* s1s    = (float*)alloc((size_t)NS * H1 * 4);
    float* s1d    = (float*)alloc((size_t)NS * H1 * 4);
    float* m1     = (float*)alloc((size_t)ND0 * H1 * 4);
    float* den1   = (float*)alloc((size_t)ND0 * H1 * 4);
    float* eb0    = (float*)alloc((size_t)E0 * H1 * 4);
    float* h1     = (float*)alloc((size_t)ND0 * F * 4);
    bf16*  h1b    = (bf16*)alloc((size_t)ND0 * F * 2);
    float* z2     = (float*)alloc((size_t)ND0 * O2 * 4);
    float* s2s    = (float*)alloc((size_t)ND0 * 4);
    float* s2d    = (float*)alloc((size_t)ND0 * 4);
    float* m2     = (float*)alloc((size_t)ND1 * 4);
    float* den2   = (float*)alloc((size_t)ND1 * 4);
    float* eb1    = (float*)alloc((size_t)E1 * 4);

    auto cdiv = [](long a, long b) -> unsigned { return (unsigned)((a + b - 1) / b); };
    const float NEG_INF = -__builtin_huge_valf();

    // ---- 1. precision casts ----
    k_cvt_bf16<<<cdiv((size_t)NS * IMGD, 256), 256, 0, stream>>>(img, Ximg, (size_t)NS * IMGD);
    k_cvt_bf16<<<cdiv((size_t)NS * BLKD, 256), 256, 0, stream>>>(blk, Xblk, (size_t)NS * BLKD);
    k_cvt_bf16<<<cdiv((size_t)F * IMGD, 256), 256, 0, stream>>>(Wimg, Wimgb, (size_t)F * IMGD);
    k_cvt_bf16<<<cdiv((size_t)F * BLKD, 256), 256, 0, stream>>>(Wblk, Wblkb, (size_t)F * BLKD);
    k_cvt_bf16<<<cdiv((size_t)F * F, 256), 256, 0, stream>>>(fc1, fc1b, (size_t)F * F);
    k_cvt_bf16<<<cdiv((size_t)O2 * F, 256), 256, 0, stream>>>(fc2, fc2b, (size_t)O2 * F);
    k_transpose_bf16<<<cdiv((size_t)F * F, 256), 256, 0, stream>>>(Wv, WvTb, F, F);
    k_transpose_bf16<<<cdiv((size_t)F * F, 256), 256, 0, stream>>>(We, WeTb, F, F);

    // ---- 2. cross-attention (4 WMMA GEMMs, gate fused into epilogues) ----
    {
        dim3 g(cdiv(F, BN), cdiv(NS, BM));
        k_gemm<1><<<g, 256, 0, stream>>>(Ximg, Wimgb, nullptr, fib, nullptr, nullptr, NS, F, IMGD);
        k_gemm<1><<<g, 256, 0, stream>>>(Xblk, Wblkb, nullptr, tib, nullptr, nullptr, NS, F, BLKD);
        k_gemm<2><<<g, 256, 0, stream>>>(fib, WvTb, fusedF, nullptr, fib, bv, NS, F, F);
        k_gemm<3><<<g, 256, 0, stream>>>(tib, WeTb, fusedF, fusedB, tib, be, NS, F, F);
        // ---- 3. layer-1 GAT projection z1 = fused @ fc1^T ----
        k_gemm<0><<<g, 256, 0, stream>>>(fusedB, fc1b, z1, nullptr, nullptr, nullptr, NS, F, F);
    }

    // ---- 4. layer-1 attention scores + edge softmax + aggregation ----
    k_scores<<<cdiv((size_t)NS * H1, 256), 256, 0, stream>>>(z1, attn1, s1s, s1d, NS, H1, O1);
    k_fill<<<cdiv((size_t)ND0 * H1, 256), 256, 0, stream>>>(m1, NEG_INF, (size_t)ND0 * H1);
    k_fill<<<cdiv((size_t)ND0 * H1, 256), 256, 0, stream>>>(den1, 0.f, (size_t)ND0 * H1);
    k_fill<<<cdiv((size_t)ND0 * F, 256), 256, 0, stream>>>(h1, 0.f, (size_t)ND0 * F);
    k_edge_max<<<cdiv((size_t)E0 * H1, 256), 256, 0, stream>>>(e0s, e0d, s1s, s1d, eb0, m1, E0, H1);
    k_edge_exp<<<cdiv((size_t)E0 * H1, 256), 256, 0, stream>>>(e0d, eb0, m1, den1, E0, H1);
    k_edge_aggr<<<E0, 256, 0, stream>>>(e0s, e0d, eb0, den1, z1, h1, E0, H1, O1);

    // ---- 5. ELU + cast ----
    k_elu_bf16<<<cdiv((size_t)ND0 * F, 256), 256, 0, stream>>>(h1, h1b, (size_t)ND0 * F);

    // ---- 6. layer-2 GAT projection z2 = h @ fc2^T ----
    {
        dim3 g(cdiv(O2, BN), cdiv(ND0, BM));
        k_gemm<0><<<g, 256, 0, stream>>>(h1b, fc2b, z2, nullptr, nullptr, nullptr, ND0, O2, F);
    }

    // ---- 7. layer-2 edge softmax + aggregation into d_out ----
    float* out = (float*)d_out;
    k_scores<<<cdiv(ND0, 256), 256, 0, stream>>>(z2, attn2, s2s, s2d, ND0, 1, O2);
    k_fill<<<cdiv(ND1, 256), 256, 0, stream>>>(m2, NEG_INF, (size_t)ND1);
    k_fill<<<cdiv(ND1, 256), 256, 0, stream>>>(den2, 0.f, (size_t)ND1);
    k_fill<<<cdiv(out_size, 256), 256, 0, stream>>>(out, 0.f, (size_t)out_size);
    k_edge_max<<<cdiv(E1, 256), 256, 0, stream>>>(e1s, e1d, s2s, s2d, eb1, m2, E1, 1);
    k_edge_exp<<<cdiv(E1, 256), 256, 0, stream>>>(e1d, eb1, m2, den2, E1, 1);
    k_edge_aggr<<<E1, 128, 0, stream>>>(e1s, e1d, eb1, den2, z2, out, E1, 1, O2);
}
